// HybridAttention_33809982554630
// MI455X (gfx1250) — matrix operand order repaired and checked by reading, not compiled, not run on hardware
//
#include <hip/hip_runtime.h>
#include <math.h>

#define N_NODESC 100000
#define N_EDGESC 400000
#define HID 128
#define HEADS 8
#define LDSTRIDE 132   // fp32 attn_feat rows: pad so rows hit distinct banks

typedef __attribute__((ext_vector_type(16))) __bf16 v16bf;
typedef __attribute__((ext_vector_type(8)))  __bf16 v8bf;
typedef __attribute__((ext_vector_type(8)))  float  v8f;

// Packed fragment layouts (pre-swizzled for v_wmma_f32_16x16x32_bf16, wave32):
//   A-frag LDS:   [kc 0..3][lane 0..31][16 bf16]     (one 32B read per frag)
//   B-frag pack:  [ct 0..7][kc 0..3][lane 0..31][16]  (one 32B read per frag)
// A layout: lanes 0-15 -> M=lane, K={0..7,16..23}; lanes 16-31 -> M=lane-16, K={8..15,24..31}
// B layout: lanes 0-15 -> N=lane, K=kk+{0..7,16..23}; lanes 16-31 -> N=lane-16, K=kk+{8..15,24..31}

__device__ __forceinline__ v8bf cvt8(const float4 a, const float4 b) {
  v8bf r;
  r[0] = (__bf16)a.x; r[1] = (__bf16)a.y; r[2] = (__bf16)a.z; r[3] = (__bf16)a.w;
  r[4] = (__bf16)b.x; r[5] = (__bf16)b.y; r[6] = (__bf16)b.z; r[7] = (__bf16)b.w;
  return r;
}

// Stage a 16x128 fp32 tile straight into bf16 A-fragment layout in LDS.
// 256 threads x 8 consecutive k-values; each 8-pack lands contiguously in one lane slot.
__device__ __forceinline__ void stage_a_bf16(const float* __restrict__ src, __bf16* aLds,
                                             int rowBase, int t) {
  const int idx = t * 8;               // 2048 floats
  const int m   = idx >> 7;
  const int cc  = idx & 127;           // multiple of 8
  const float* g = src + (size_t)(rowBase + m) * HID + cc;
  const float4 v0 = *(const float4*)(g);
  const float4 v1 = *(const float4*)(g + 4);
  const int kc   = cc >> 5;
  const int lane = (m & 15) + (((cc >> 3) & 1) << 4);
  const int ib   = ((cc >> 4) & 1) << 3;
  *(v8bf*)(aLds + ((kc * 32 + lane) * 16 + ib)) = cvt8(v0, v1);
}

__device__ __forceinline__ v8f wmma_gemm_pk(const v16bf a[4], const __bf16* __restrict__ Wpk,
                                            int lane, int ct) {
  v8f c = {};
#pragma unroll
  for (int kc = 0; kc < 4; ++kc) {
    const v16bf b = *(const v16bf*)(Wpk + (size_t)(((ct * 4 + kc) * 32) + lane) * 16);
    c = __builtin_amdgcn_wmma_f32_16x16x32_bf16(false, a[kc], false, b, (short)0, c, false, false);
  }
  return c;
}

// ---- one-shot: pack 8 fp32 128x128 weights into bf16 B-fragment layout -
__global__ void pack_weights_kernel(
    const float* __restrict__ W0, const float* __restrict__ W1,
    const float* __restrict__ W2, const float* __restrict__ W3,
    const float* __restrict__ W4, const float* __restrict__ W5,
    const float* __restrict__ W6, const float* __restrict__ W7,
    __bf16* __restrict__ out) {
  const int id = blockIdx.x * blockDim.x + threadIdx.x;   // 8192 total
  if (id >= 8 * 8 * 4 * 32) return;
  const int mat  = id >> 10;
  const int rest = id & 1023;
  const int ct   = rest >> 7;
  const int kc   = (rest >> 5) & 3;
  const int lane = rest & 31;
  const float* W;
  switch (mat) {
    case 0: W = W0; break; case 1: W = W1; break; case 2: W = W2; break;
    case 3: W = W3; break; case 4: W = W4; break; case 5: W = W5; break;
    case 6: W = W6; break; default: W = W7; break;
  }
  const int n  = ct * 16 + (lane & 15);
  const int ko = (lane & 16) ? 8 : 0;
  const float* p = W + (size_t)(kc * 32 + ko) * HID + n;
  v16bf b;
#pragma unroll
  for (int i = 0; i < 16; ++i) b[i] = (__bf16)p[(size_t)(i + (i >= 8 ? 8 : 0)) * HID];
  *(v16bf*)(out + (size_t)id * 16) = b;
}

// ---- kernel 0: per-call scratch init -----------------------------------
__global__ void init_kernel(float* __restrict__ agg, float* __restrict__ segsum,
                            int* __restrict__ segmax) {
  const int i = blockIdx.x * blockDim.x + threadIdx.x;
  if (i < N_NODESC * HID) agg[i] = 0.0f;
  if (i < N_NODESC * HEADS) { segsum[i] = 0.0f; segmax[i] = (int)0xFF800000u; /* -inf */ }
}

// ---- kernel 1: Q,K,V = x @ W + b (3 fused GEMMs, one x-tile stage) -----
__global__ __launch_bounds__(256) void node_qkv_kernel(
    const float* __restrict__ x,
    const __bf16* __restrict__ Wq, const float* __restrict__ bq,
    const __bf16* __restrict__ Wk, const float* __restrict__ bk,
    const __bf16* __restrict__ Wv, const float* __restrict__ bv,
    float* __restrict__ Qo, float* __restrict__ Ko, float* __restrict__ Vo) {
  __shared__ __bf16 aLds[4 * 32 * 16];
  const int rowBase = blockIdx.x * 16;
  const int t = threadIdx.x;
  stage_a_bf16(x, aLds, rowBase, t);
  __syncthreads();
  const int lane = t & 31;
  const int ct   = t >> 5;
  v16bf a[4];
#pragma unroll
  for (int kc = 0; kc < 4; ++kc) a[kc] = *(const v16bf*)(aLds + (kc * 32 + lane) * 16);
  const v8f cq = wmma_gemm_pk(a, Wq, lane, ct);
  const v8f ck = wmma_gemm_pk(a, Wk, lane, ct);
  const v8f cv = wmma_gemm_pk(a, Wv, lane, ct);
  const int ncol = ct * 16 + (lane & 15);
  const int mb   = (lane & 16) ? 8 : 0;
  const float bq_ = bq[ncol], bk_ = bk[ncol], bv_ = bv[ncol];
#pragma unroll
  for (int r = 0; r < 8; ++r) {
    const size_t o = (size_t)(rowBase + r + mb) * HID + ncol;
    Qo[o] = cq[r] + bq_;
    Ko[o] = ck[r] + bk_;
    Vo[o] = cv[r] + bv_;
  }
}

// ---- kernel 2: edge mega-kernel ----------------------------------------
// Ew,Eb,Ev GEMMs -> signed_sqrt/ReLU into LDS (fp32 rows + bf16 A-frags)
// -> fused Eo GEMM -> scores (128->8) + float atomic-max into segmax.
__global__ __launch_bounds__(256) void edge_mega_kernel(
    const float* __restrict__ edge_attr, const int* __restrict__ edge_index,
    const __bf16* __restrict__ Wew, const float* __restrict__ bew,
    const __bf16* __restrict__ Web, const float* __restrict__ beb,
    const __bf16* __restrict__ Wev, const float* __restrict__ bev,
    const __bf16* __restrict__ Weo, const float* __restrict__ beo,
    const float* __restrict__ WA,   const float* __restrict__ bA,
    const float* __restrict__ Q,    const float* __restrict__ K,
    float* __restrict__ Evo, float* __restrict__ scores,
    int* __restrict__ segmax, float* __restrict__ eout) {
  __shared__ __bf16 aLds[4 * 32 * 16];
  __shared__ __bf16 afbf[4 * 32 * 16];
  __shared__ float  af[16 * LDSTRIDE];
  __shared__ int s_src[16], s_tgt[16];
  const int eBase = blockIdx.x * 16;
  const int t = threadIdx.x;
  if (t < 16) {
    s_src[t] = edge_index[eBase + t];
    s_tgt[t] = edge_index[N_EDGESC + eBase + t];
  }
  stage_a_bf16(edge_attr, aLds, eBase, t);
  __syncthreads();
  const int lane = t & 31;
  const int ct   = t >> 5;
  v16bf a[4];
#pragma unroll
  for (int kc = 0; kc < 4; ++kc) a[kc] = *(const v16bf*)(aLds + (kc * 32 + lane) * 16);
  const v8f cw = wmma_gemm_pk(a, Wew, lane, ct);
  const v8f cb = wmma_gemm_pk(a, Web, lane, ct);
  const v8f cv = wmma_gemm_pk(a, Wev, lane, ct);
  const int ncol = ct * 16 + (lane & 15);
  const int mb   = (lane & 16) ? 8 : 0;
  const float bw = bew[ncol], bb = beb[ncol], bv_ = bev[ncol];
  // attn_feat bf16-fragment coords for column ncol (same for all 8 rows)
  const int fkc = ncol >> 5;
  const int fii = (ncol & 7) + (((ncol >> 4) & 1) << 3);
  const int flb = ((ncol >> 3) & 1) << 4;
#pragma unroll
  for (int r = 0; r < 8; ++r) {
    const int m = r + mb;
    const int e = eBase + m;
    const int sn = s_src[m], tn = s_tgt[m];
    const float ew = cw[r] + bw;
    const float eb = cb[r] + bb;
    Evo[(size_t)e * HID + ncol] = cv[r] + bv_;
    const float z  = (Q[(size_t)sn * HID + ncol] + K[(size_t)tn * HID + ncol]) * ew;
    const float sg = (z > 0.0f) ? 1.0f : ((z < 0.0f) ? -1.0f : 0.0f);
    const float v  = fmaxf(sg * sqrtf(fabsf(z) + 1e-8f) + eb, 0.0f);
    af[m * LDSTRIDE + ncol] = v;
    afbf[(fkc * 32 + ((m & 15) + flb)) * 16 + fii] = (__bf16)v;
  }
  __syncthreads();
  // edge_attr_out = attn_feat @ W_Eo + b_Eo, fragments straight from LDS
  v16bf f[4];
#pragma unroll
  for (int kc = 0; kc < 4; ++kc) f[kc] = *(const v16bf*)(afbf + (kc * 32 + lane) * 16);
  const v8f co = wmma_gemm_pk(f, Weo, lane, ct);
  const float bo = beo[ncol];
#pragma unroll
  for (int r = 0; r < 8; ++r)
    eout[(size_t)(eBase + r + mb) * HID + ncol] = co[r] + bo;
  // scores: 16 edges x 8 heads, one thread each
  if (t < 128) {
    const int r = t >> 3, j = t & 7;
    const float* row = af + r * LDSTRIDE;
    float s = bA[j];
#pragma unroll 8
    for (int i = 0; i < HID; ++i) s += row[i] * WA[i * HEADS + j];
    scores[(size_t)(eBase + r) * HEADS + j] = s;
    int* addr = segmax + s_tgt[r] * HEADS + j;
    if (s >= 0.0f) atomicMax(addr, __float_as_int(s));
    else           atomicMin((unsigned int*)addr, (unsigned int)__float_as_int(s));
  }
}

// ---- kernel 3: ex = exp(score - seg_max[tgt]); segsum += ex ------------
__global__ void softmax_exp_kernel(const float* __restrict__ scores,
                                   const int* __restrict__ segmax,
                                   const int* __restrict__ edge_index,
                                   float* __restrict__ ex, float* __restrict__ segsum) {
  const int i = blockIdx.x * blockDim.x + threadIdx.x;
  if (i >= N_EDGESC * HEADS) return;
  const int e = i >> 3, j = i & 7;
  const int tn = edge_index[N_EDGESC + e];
  const float m = __int_as_float(segmax[tn * HEADS + j]);
  const float v = expf(scores[i] - m);
  ex[i] = v;
  atomicAdd(&segsum[tn * HEADS + j], v);
}

// ---- kernel 4: agg[tgt] += (V[src] + Ev) * alpha -----------------------
__global__ void message_kernel(const float* __restrict__ Vn, const float* __restrict__ Evo,
                               const float* __restrict__ ex, const float* __restrict__ segsum,
                               const int* __restrict__ edge_index, float* __restrict__ agg) {
  const int i = blockIdx.x * blockDim.x + threadIdx.x;
  if (i >= N_EDGESC * 32) return;
  const int e = i >> 5;
  const int f = (i & 31) << 2;          // 4 floats, all within one head
  const int sn = edge_index[e];
  const int tn = edge_index[N_EDGESC + e];
  const int h = f >> 4;
  const float alpha = ex[(size_t)e * HEADS + h] / (segsum[tn * HEADS + h] + 1e-16f);
  const float4 v = *(const float4*)(Vn  + (size_t)sn * HID + f);
  const float4 w = *(const float4*)(Evo + (size_t)e  * HID + f);
  float* dst = agg + (size_t)tn * HID + f;
  atomicAdd(dst + 0, (v.x + w.x) * alpha);
  atomicAdd(dst + 1, (v.y + w.y) * alpha);
  atomicAdd(dst + 2, (v.z + w.z) * alpha);
  atomicAdd(dst + 3, (v.w + w.w) * alpha);
}

// ---- kernel 5: x_out = agg @ W_O + b_O ---------------------------------
__global__ __launch_bounds__(256) void node_out_kernel(
    const float* __restrict__ agg, const __bf16* __restrict__ Wo,
    const float* __restrict__ bo, float* __restrict__ xout) {
  __shared__ __bf16 aLds[4 * 32 * 16];
  const int rowBase = blockIdx.x * 16;
  const int t = threadIdx.x;
  stage_a_bf16(agg, aLds, rowBase, t);
  __syncthreads();
  const int lane = t & 31;
  const int ct   = t >> 5;
  v16bf a[4];
#pragma unroll
  for (int kc = 0; kc < 4; ++kc) a[kc] = *(const v16bf*)(aLds + (kc * 32 + lane) * 16);
  const v8f c = wmma_gemm_pk(a, Wo, lane, ct);
  const int ncol = ct * 16 + (lane & 15);
  const int mb   = (lane & 16) ? 8 : 0;
  const float b = bo[ncol];
#pragma unroll
  for (int r = 0; r < 8; ++r)
    xout[(size_t)(rowBase + r + mb) * HID + ncol] = c[r] + b;
}

extern "C" void kernel_launch(void* const* d_in, const int* in_sizes, int n_in,
                              void* d_out, int out_size, void* d_ws, size_t ws_size,
                              hipStream_t stream) {
  const float* x         = (const float*)d_in[0];
  const float* edge_attr = (const float*)d_in[1];
  const int*   edge_idx  = (const int*)d_in[2];
  // d_in[3] = batch (unused: single graph)
  const float* W_Q  = (const float*)d_in[4];  const float* b_Q  = (const float*)d_in[5];
  const float* W_K  = (const float*)d_in[6];  const float* b_K  = (const float*)d_in[7];
  const float* W_V  = (const float*)d_in[8];  const float* b_V  = (const float*)d_in[9];
  const float* W_Ew = (const float*)d_in[10]; const float* b_Ew = (const float*)d_in[11];
  const float* W_Eb = (const float*)d_in[12]; const float* b_Eb = (const float*)d_in[13];
  const float* W_Ev = (const float*)d_in[14]; const float* b_Ev = (const float*)d_in[15];
  const float* W_O  = (const float*)d_in[16]; const float* b_O  = (const float*)d_in[17];
  const float* W_Eo = (const float*)d_in[18]; const float* b_Eo = (const float*)d_in[19];
  const float* W_A  = (const float*)d_in[20]; const float* b_A  = (const float*)d_in[21];

  // workspace carve-up (float units)
  float* ws = (float*)d_ws;
  size_t off = 0;
  float* Q      = ws + off; off += (size_t)N_NODESC * HID;
  float* K      = ws + off; off += (size_t)N_NODESC * HID;
  float* V      = ws + off; off += (size_t)N_NODESC * HID;
  float* Evo    = ws + off; off += (size_t)N_EDGESC * HID;
  float* scores = ws + off; off += (size_t)N_EDGESC * HEADS;
  float* ex     = ws + off; off += (size_t)N_EDGESC * HEADS;
  int*   segmax = (int*)(ws + off); off += (size_t)N_NODESC * HEADS;
  float* segsum = ws + off; off += (size_t)N_NODESC * HEADS;
  float* agg    = ws + off; off += (size_t)N_NODESC * HID;
  __bf16* Wpk   = (__bf16*)(ws + off); off += (size_t)8 * 16384 / 2;  // 8 packed matrices

  const size_t WPK = 16384;  // bf16 elements per packed matrix
  __bf16* pWq  = Wpk + 0 * WPK;
  __bf16* pWk  = Wpk + 1 * WPK;
  __bf16* pWv  = Wpk + 2 * WPK;
  __bf16* pWew = Wpk + 3 * WPK;
  __bf16* pWeb = Wpk + 4 * WPK;
  __bf16* pWev = Wpk + 5 * WPK;
  __bf16* pWeo = Wpk + 6 * WPK;
  __bf16* pWo  = Wpk + 7 * WPK;

  float* x_out = (float*)d_out;
  float* e_out = (float*)d_out + (size_t)N_NODESC * HID;

  pack_weights_kernel<<<32, 256, 0, stream>>>(
      W_Q, W_K, W_V, W_Ew, W_Eb, W_Ev, W_Eo, W_O, Wpk);

  init_kernel<<<(N_NODESC * HID + 255) / 256, 256, 0, stream>>>(agg, segsum, segmax);

  node_qkv_kernel<<<N_NODESC / 16, 256, 0, stream>>>(
      x, pWq, b_Q, pWk, b_K, pWv, b_V, Q, K, V);

  edge_mega_kernel<<<N_EDGESC / 16, 256, 0, stream>>>(
      edge_attr, edge_idx,
      pWew, b_Ew, pWeb, b_Eb, pWev, b_Ev, pWeo, b_Eo, W_A, b_A,
      Q, K, Evo, scores, segmax, e_out);

  softmax_exp_kernel<<<(N_EDGESC * HEADS + 255) / 256, 256, 0, stream>>>(
      scores, segmax, edge_idx, ex, segsum);

  message_kernel<<<(N_EDGESC * 32 + 255) / 256, 256, 0, stream>>>(
      V, Evo, ex, segsum, edge_idx, agg);

  node_out_kernel<<<N_NODESC / 16, 256, 0, stream>>>(agg, pWo, b_O, x_out);
}
